// GaussianRadialBasisLayer_13838384628107
// MI455X (gfx1250) — matrix operand order repaired and checked by reading, not compiled, not run on hardware
//
#include <hip/hip_runtime.h>

typedef __attribute__((ext_vector_type(16))) _Float16 v16h;
typedef __attribute__((ext_vector_type(8)))  float    v8f;

#define N_ROWS  131072
#define DDIM    256
#define UDIM    512
#define KB      8            // k-blocks of 32 over D=256 (each drives TWO wmmas)
#define LN2PI_2 0.9189385332046727f

// ---------------------------------------------------------------------------
// Pack B (512 x 512 f16) into WMMA B-fragment layout.
//   Combined K space: K<256 -> coeff for x^2 : -0.5/s^2
//                     K>=256 -> coeff for x  :  l/s^2
//   chunk t = ((uT*16 + kb)*32 + lane); lane<16 -> col=lane,   K=kbase+0..15
//                                       lane>=16 -> col=lane-16, K=kbase+16..31
// ---------------------------------------------------------------------------
__global__ __launch_bounds__(256) void pack_b_kernel(const float* __restrict__ l,
                                                     const float* __restrict__ s,
                                                     _Float16* __restrict__ Bp) {
  unsigned t    = blockIdx.x * blockDim.x + threadIdx.x;   // 0 .. 32*16*32-1
  unsigned lane = t & 31u;
  unsigned kb   = (t >> 5) & 15u;
  unsigned uT   = t >> 9;
  unsigned c    = lane & 15u;
  unsigned u    = uT * 16u + c;
  unsigned kbase = kb * 32u + (lane >> 4) * 16u;
  bool     firstHalf = (kb < 8u);                          // uniform per wave
  unsigned dbase = firstHalf ? kbase : (kbase - 256u);

  v16h o;
#pragma unroll
  for (int j = 0; j < 16; ++j) {
    unsigned d  = dbase + j;
    float sv    = s[(size_t)d * UDIM + u];
    float is2   = 1.0f / (sv * sv);
    float v     = firstHalf ? (-0.5f * is2) : (l[(size_t)d * UDIM + u] * is2);
    o[j] = (_Float16)v;
  }
  *(v16h*)(Bp + (size_t)t * 16u) = o;
}

// ---------------------------------------------------------------------------
// bias[u] = -0.5*sum_d l^2/s^2 - 0.5*sum_d log(s^2) - D*ln(2pi)/2   (f32)
// ---------------------------------------------------------------------------
__global__ __launch_bounds__(256) void bias_kernel(const float* __restrict__ l,
                                                   const float* __restrict__ s,
                                                   float* __restrict__ bias) {
  unsigned u = blockIdx.x * blockDim.x + threadIdx.x;
  if (u >= UDIM) return;
  float q = 0.0f, ld = 0.0f;
  for (int d = 0; d < DDIM; ++d) {
    float lv = l[(size_t)d * UDIM + u];
    float sv = s[(size_t)d * UDIM + u];
    float s2 = sv * sv;
    q  += lv * lv / s2;
    ld += __logf(s2);
  }
  bias[u] = -0.5f * q - 0.5f * ld - (float)DDIM * LN2PI_2;
}

// ---------------------------------------------------------------------------
// Fused GEMM: out[n,u] = sum_d x^2*(-0.5/s^2) + x*(l/s^2) + bias[u]
// 128x128 block tile, 8 waves (2M x 4N), wave = 4x2 16x16 frags.
// A fragments built on the fly from f32 x (load once, cvt to f16, square in
// f16) -> two WMMAs per A fragment against the two B coefficient regions.
// x stays L2-resident across the 4 column blocks (grid.x = column dim).
// ---------------------------------------------------------------------------
__global__ __launch_bounds__(256) void gemm_kernel(const float* __restrict__ x,
                                                   const _Float16* __restrict__ Bp,
                                                   const float* __restrict__ bias,
                                                   float* __restrict__ out) {
  const unsigned lane = threadIdx.x & 31u;
  const unsigned w    = threadIdx.x >> 5;
  const unsigned wm   = w & 1u;          // 0..1  -> 64-row half
  const unsigned wn   = w >> 1;          // 0..3  -> 32-col quarter
  const unsigned m    = lane & 15u;
  const unsigned hi8  = (lane >> 4) * 16u;  // A-layout K offset for lanes 16..31

  v8f acc[4][2];
#pragma unroll
  for (int mi = 0; mi < 4; ++mi)
#pragma unroll
    for (int ni = 0; ni < 2; ++ni) { v8f z = {}; acc[mi][ni] = z; }

  // Per-mi x row pointers (wave tile rows = blockIdx.y*128 + wm*64 + mi*16 + m)
  const float* xr[4];
#pragma unroll
  for (int mi = 0; mi < 4; ++mi) {
    unsigned row = blockIdx.y * 128u + wm * 64u + mi * 16u + m;
    xr[mi] = x + (size_t)row * DDIM;
  }
  // Per-ni packed-B fragment base pointers (chunk = ((uT*16 + kb)*32+lane)*16)
  const _Float16* bp[2];
#pragma unroll
  for (int ni = 0; ni < 2; ++ni) {
    unsigned uT = blockIdx.x * 8u + wn * 2u + ni;
    bp[ni] = Bp + ((size_t)uT * 16u * 32u + lane) * 16u;
  }

  for (int kb = 0; kb < KB; ++kb) {
    // B fragments: kb -> x^2 coefficients, kb+8 -> x coefficients
    v16h bl[2], bh[2];
#pragma unroll
    for (int ni = 0; ni < 2; ++ni) {
      bl[ni] = *(const v16h*)(bp[ni] + (size_t)kb * 512u);
      bh[ni] = *(const v16h*)(bp[ni] + (size_t)kb * 512u + 4096u);
    }
    // A fragments from f32 x: K cols {g0..g0+15}
    unsigned g0 = kb * 32u + hi8;
    v16h ax[4], sq[4];
#pragma unroll
    for (int mi = 0; mi < 4; ++mi) {
      float4 p0 = *(const float4*)(xr[mi] + g0);
      float4 p1 = *(const float4*)(xr[mi] + g0 + 4);
      float4 p2 = *(const float4*)(xr[mi] + g0 + 8);
      float4 p3 = *(const float4*)(xr[mi] + g0 + 12);
      v16h a;
      a[0]  = (_Float16)p0.x; a[1]  = (_Float16)p0.y;
      a[2]  = (_Float16)p0.z; a[3]  = (_Float16)p0.w;
      a[4]  = (_Float16)p1.x; a[5]  = (_Float16)p1.y;
      a[6]  = (_Float16)p1.z; a[7]  = (_Float16)p1.w;
      a[8]  = (_Float16)p2.x; a[9]  = (_Float16)p2.y;
      a[10] = (_Float16)p2.z; a[11] = (_Float16)p2.w;
      a[12] = (_Float16)p3.x; a[13] = (_Float16)p3.y;
      a[14] = (_Float16)p3.z; a[15] = (_Float16)p3.w;
      ax[mi] = a;
      sq[mi] = a * a;                       // v_pk_mul_f16, x^2 in f16
    }
    // 16 WMMAs: x^2-part then x-part accumulate into the same tile
#pragma unroll
    for (int mi = 0; mi < 4; ++mi)
#pragma unroll
      for (int ni = 0; ni < 2; ++ni) {
        acc[mi][ni] = __builtin_amdgcn_wmma_f32_16x16x32_f16(
            false, sq[mi], false, bl[ni], (short)0, acc[mi][ni], false, false);
        acc[mi][ni] = __builtin_amdgcn_wmma_f32_16x16x32_f16(
            false, ax[mi], false, bh[ni], (short)0, acc[mi][ni], false, false);
      }
  }

  // Epilogue: C frag: lane<16 -> N=lane, M=r ; lane>=16 -> N=lane-16, M=r+8.
  // Non-temporal stores: out is write-once; keep x resident in L2.
  unsigned colBase = blockIdx.x * 128u + wn * 32u;
  unsigned rowBase = blockIdx.y * 128u + wm * 64u + (lane >> 4) * 8u;
#pragma unroll
  for (int ni = 0; ni < 2; ++ni) {
    unsigned col = colBase + ni * 16u + m;
    float bv = bias[col];
#pragma unroll
    for (int mi = 0; mi < 4; ++mi) {
      unsigned row0 = rowBase + mi * 16u;
#pragma unroll
      for (int r = 0; r < 8; ++r)
        __builtin_nontemporal_store(acc[mi][ni][r] + bv,
                                    &out[(size_t)(row0 + r) * UDIM + col]);
    }
  }
}

// ---------------------------------------------------------------------------
extern "C" void kernel_launch(void* const* d_in, const int* in_sizes, int n_in,
                              void* d_out, int out_size, void* d_ws, size_t ws_size,
                              hipStream_t stream) {
  const float* x = (const float*)d_in[0];   // [N, D] f32
  const float* l = (const float*)d_in[1];   // [D, U] f32
  const float* s = (const float*)d_in[2];   // [D, U] f32
  float* out = (float*)d_out;               // [N, U] f32

  _Float16* Bp  = (_Float16*)d_ws;                               // 512 KB
  float*    bia = (float*)((char*)d_ws + (size_t)512 * UDIM * 2); // 2 KB

  pack_b_kernel<<<(32 * 16 * 32) / 256, 256, 0, stream>>>(l, s, Bp);
  bias_kernel<<<UDIM / 256, 256, 0, stream>>>(l, s, bia);

  // grid.x = column blocks (4) varies fastest -> the 4 blocks sharing one
  // 128-row x slab are adjacent in dispatch order => x read from HBM once.
  dim3 grid(UDIM / 128, N_ROWS / 128);
  gemm_kernel<<<grid, 256, 0, stream>>>(x, Bp, bia, out);
}
